// ScaleDotProductionAttention_3143916061018
// MI455X (gfx1250) — matrix operand order repaired and checked by reading, not compiled, not run on hardware
//
#include <hip/hip_runtime.h>
#include <hip/hip_bf16.h>

typedef _Float16 v16h __attribute__((ext_vector_type(16)));
typedef float    v8f  __attribute__((ext_vector_type(8)));
typedef int      v4i32 __attribute__((vector_size(16)));

#define AS1 __attribute__((address_space(1)))
#define AS3 __attribute__((address_space(3)))

#define B_ 4
#define H_ 16
#define S_ 2048
#define D_ 64
#define CT (S_/16)          // 128 column (key) tiles
#define SCALE 0.125f        // 1/sqrt(64)

// Dynamic LDS layout (bytes):
//   Pbuf   [16][2048] f32      : 131072
//   Qh     [16][64]   f16      :   2048
//   red    [16][16]   f32      :   1024
//   rowmax [16], rowsum[16]    :    128
//   Obuf   [2][16][64] f32     :   8192
//   Kstage [8 waves][2][16][64]: 65536   (f32, async double-buffer)
#define KSTAGE_OFF (131072 + 2048 + 1024 + 128 + 8192)
#define SMEM_BYTES (KSTAGE_OFF + 65536)

// ---- gfx1250 async global->LDS copy (ASYNCcnt path), with safe fallback ----
#if defined(__AMDGCN__) && __has_builtin(__builtin_amdgcn_global_load_async_to_lds_b128)
#define HAVE_ASYNC_LDS 1
#else
#define HAVE_ASYNC_LDS 0
#endif

#if defined(__AMDGCN__) && __has_builtin(__builtin_amdgcn_s_wait_asynccnt)
#define WAIT_ASYNC(n) __builtin_amdgcn_s_wait_asynccnt(n)
#elif defined(__AMDGCN__)
#define WAIT_ASYNC(n) asm volatile("s_wait_asynccnt %0" :: "i"(n) : "memory")
#else
#define WAIT_ASYNC(n)
#endif

// Copy one 16x64-float K tile (4 KB) into LDS: 8 x (32 lanes x 16B) = 8 async ops.
__device__ __forceinline__ void stage_k_tile(const float* __restrict__ gsrc,
                                             float* lds_dst, int lane) {
#if HAVE_ASYNC_LDS
    const char* g = (const char*)gsrc + lane * 16;
    char*       l = (char*)lds_dst + lane * 16;
    #pragma unroll
    for (int i = 0; i < 8; ++i) {
        __builtin_amdgcn_global_load_async_to_lds_b128(
            (AS1 v4i32*)(g + i * 512),
            (AS3 v4i32*)(l + i * 512),
            0, 0);
    }
#else
    const float4* s = (const float4*)((const char*)gsrc + lane * 16);
    float4*       d = (float4*)((char*)lds_dst + lane * 16);
    #pragma unroll
    for (int i = 0; i < 8; ++i) d[i * 32] = s[i * 32];
#endif
}

__global__ __launch_bounds__(256) void sdpa_fused_kernel(
    const float* __restrict__ q, const float* __restrict__ k,
    const float* __restrict__ v, const int* __restrict__ mask,
    float* __restrict__ out_o, float* __restrict__ out_s)
{
    extern __shared__ char smem[];
    float*    Pbuf   = (float*)smem;                                  // [16][S_]
    _Float16* Qh     = (_Float16*)(smem + 16*S_*4);                   // [16][D_]
    float*    red    = (float*)(smem + 16*S_*4 + 16*D_*2);            // [16][16]
    float*    rowmax = red + 256;
    float*    rowsum = rowmax + 16;
    float*    Obuf   = rowsum + 16;                                   // [2][16][D_]
    float*    Kstage = (float*)(smem + KSTAGE_OFF);                   // [8][2][16][D_]

    const int tid  = threadIdx.x;
    const int wave = tid >> 5;
    const int lane = tid & 31;

    const int bh    = blockIdx.x / CT;
    const int qtile = blockIdx.x % CT;
    const int b     = bh / H_;

    const float* qbase = q + ((size_t)bh*S_ + qtile*16) * D_;
    const float* kbase = k + (size_t)bh*S_*D_;
    const float* vbase = v + (size_t)bh*S_*D_;
    const int*   mbase = mask + (size_t)b*S_*S_ + (size_t)(qtile*16)*S_;
    float* obase = out_o + ((size_t)bh*S_ + qtile*16) * D_;
    float* sbase = out_s + ((size_t)bh*S_ + qtile*16) * S_;

    // ---- Phase 0: stage Q tile into LDS as f16 (16x64 = 1024 elems) ----
    for (int i = tid; i < 16*D_; i += 256) Qh[i] = (_Float16)qbase[i];
    __syncthreads();

    // WMMA operand lane mapping (per CDNA5 ISA 7.12.2, wave32):
    //  A (16x32 f16): lane -> M = lane&15 ; K offset (lane>>4)*8, halves at K and K+16
    //  B (32x16 f16): lane -> N = lane&15 ; K offset (lane>>4)*8, halves at K and K+16
    //  C (16x16 f32): vgpr r, lane -> M = r + 8*(lane>>4), N = lane&15
    const int m     = lane & 15;
    const int abase = (lane >> 4) * 8;
    const int bbase = (lane >> 4) * 16;
    const int Moff  = (lane >> 4) * 8;

    // A operands for Q (invariant over key tiles): K dim = D = 64 -> two 16x32 chunks
    v16h a0, a1;
    #pragma unroll
    for (int t = 0; t < 8; ++t) {
        a0[t]   = Qh[m*D_ +      abase + t];
        a0[8+t] = Qh[m*D_ + 16 + abase + t];
        a1[t]   = Qh[m*D_ + 32 + abase + t];
        a1[8+t] = Qh[m*D_ + 48 + abase + t];
    }

    // ---- Phase 1: S = (Q K^T) * scale, mask -> Pbuf (async double-buffered K) ----
    {
        float* Kst = Kstage + wave * 2 * (16*D_);   // this wave's two 4 KB buffers
        stage_k_tile(kbase + (size_t)(wave*16)*D_, Kst, lane);   // preload tile j=wave
        int cur = 0;
        for (int j = wave; j < CT; j += 8) {
            const int key0 = j * 16;
            if (j + 8 < CT) {
                stage_k_tile(kbase + (size_t)((j+8)*16)*D_, Kst + (cur^1)*(16*D_), lane);
                WAIT_ASYNC(8);          // previous 8 async ops (tile j) complete
            } else {
                WAIT_ASYNC(0);
            }
#if defined(__AMDGCN__)
            asm volatile("" ::: "memory");
#endif
            const float* krow = Kst + cur*(16*D_) + m*D_;  // staged row n = m

            v16h b0, b1;
            #pragma unroll
            for (int t = 0; t < 16; ++t) {
                b0[t] = (_Float16)krow[     abase + t + (t & 8)];
                b1[t] = (_Float16)krow[32 + abase + t + (t & 8)];
            }
            v8f acc = {};
            acc = __builtin_amdgcn_wmma_f32_16x16x32_f16(false, a0, false, b0, (short)0, acc, false, false);
            acc = __builtin_amdgcn_wmma_f32_16x16x32_f16(false, a1, false, b1, (short)0, acc, false, false);

            #pragma unroll
            for (int r = 0; r < 8; ++r) {
                const int row = r + Moff;
                float val = acc[r] * SCALE;
                if (mbase[row*S_ + key0 + m] == 0) val = 1e-12f;  // faithful: fill, not -inf
                Pbuf[row*S_ + key0 + m] = val;
            }
            cur ^= 1;
        }
    }
    __syncthreads();

    // ---- Phase 2: row softmax over Pbuf; write normalized score to global ----
    {
        const int row = tid >> 4;   // 0..15
        const int sub = tid & 15;   // 16 threads cooperate per row

        float mx = -3.402823466e38f;
        for (int c = sub; c < S_; c += 16) mx = fmaxf(mx, Pbuf[row*S_ + c]);
        red[row*16 + sub] = mx;
        __syncthreads();
        if (tid < 16) {
            float r0 = red[tid*16];
            #pragma unroll
            for (int i2 = 1; i2 < 16; ++i2) r0 = fmaxf(r0, red[tid*16 + i2]);
            rowmax[tid] = r0;
        }
        __syncthreads();

        const float rm = rowmax[row];
        float sum = 0.f;
        for (int c = sub; c < S_; c += 16) {
            float e = __expf(Pbuf[row*S_ + c] - rm);
            Pbuf[row*S_ + c] = e;
            sum += e;
        }
        red[row*16 + sub] = sum;
        __syncthreads();
        if (tid < 16) {
            float s0 = 0.f;
            #pragma unroll
            for (int i2 = 0; i2 < 16; ++i2) s0 += red[tid*16 + i2];
            rowsum[tid] = s0;
        }
        __syncthreads();

        const float inv = 1.0f / rowsum[row];
        for (int c = sub; c < S_; c += 16) {
            float p = Pbuf[row*S_ + c] * inv;
            Pbuf[row*S_ + c] = p;
            sbase[(size_t)row*S_ + c] = p;   // mandatory 1 GB score write, done once
        }
    }
    __syncthreads();

    // ---- Phase 3: O = P @ V via WMMA; 8 waves = 4 N-tiles x 2 K-halves ----
    {
        const int ntile = wave & 3;
        const int khalf = wave >> 2;
        const int n0 = ntile * 16;

        v8f acco = {};
        for (int jj = khalf*64; jj < khalf*64 + 64; jj += 2) {
            const int c0 = jj * 16;              // 32 P-columns / V-rows: c0..c0+31
            if (jj + 2 < khalf*64 + 64)          // prefetch next chunk's V rows
                __builtin_prefetch(vbase + (size_t)((jj+2)*16 + bbase)*D_ + n0 + m, 0, 3);

            v16h pa, vb;
            #pragma unroll
            for (int t = 0; t < 8; ++t) {
                pa[t]   = (_Float16)Pbuf[m*S_ + c0 +      abase + t];
                pa[8+t] = (_Float16)Pbuf[m*S_ + c0 + 16 + abase + t];
            }
            #pragma unroll
            for (int t2 = 0; t2 < 16; ++t2) {
                vb[t2] = (_Float16)vbase[(size_t)(c0 + abase + t2 + (t2 & 8))*D_ + n0 + m];
            }
            acco = __builtin_amdgcn_wmma_f32_16x16x32_f16(false, pa, false, vb, (short)0, acco, false, false);
        }
        #pragma unroll
        for (int r = 0; r < 8; ++r) {
            Obuf[(khalf*16 + (r + Moff))*D_ + n0 + m] = acco[r];
        }
    }
    __syncthreads();

    // ---- Final: reduce the two K-halves, write 16x64 output tile ----
    for (int i = tid; i < 16*D_; i += 256) {
        obase[i] = Obuf[i] + Obuf[16*D_ + i];
    }
}

extern "C" void kernel_launch(void* const* d_in, const int* in_sizes, int n_in,
                              void* d_out, int out_size, void* d_ws, size_t ws_size,
                              hipStream_t stream) {
    const float* q    = (const float*)d_in[0];
    const float* k    = (const float*)d_in[1];
    const float* v    = (const float*)d_in[2];
    const int*   mask = (const int*)d_in[3];

    float* out   = (float*)d_out;
    float* out_o = out;                                    // [B,H,S,D]
    float* out_s = out + (size_t)B_*H_*S_*D_;              // [B,H,S,S]

    dim3 grid(B_*H_*CT);   // 8192 workgroups, one per (b,h,q-tile)
    dim3 block(256);       // 8 wave32
    sdpa_fused_kernel<<<grid, block, SMEM_BYTES, stream>>>(q, k, v, mask, out_o, out_s);
}
